// MambaOnlyLM_61512521614003
// MI455X (gfx1250) — matrix operand order repaired and checked by reading, not compiled, not run on hardware
//
#include <hip/hip_runtime.h>

// ---------------- model dims ----------------
#define BATCH   2
#define SEQ     2048
#define T_TOK   (BATCH * SEQ)       // 4096 tokens
#define DM      768                 // d_model
#define DI      1536                // d_inner
#define DS_N    128                 // d_state
#define NH      24                  // heads
#define HD      64                  // head dim
#define DCONV   4
#define CD      (DI + 2 * DS_N)     // 1792 conv dim
#define DIP     (2 * DI + 2 * DS_N + NH) // 3352 in_proj out
#define FF      2048
#define VOCAB   32000
#define NLAYERS 12
#define EPS_F   1e-5f

typedef __attribute__((ext_vector_type(16))) __bf16 bf16x16;
typedef __attribute__((ext_vector_type(8)))  float  f32x8;
typedef __attribute__((ext_vector_type(4)))  int    i32x4;

#define AS_GLOBAL __attribute__((address_space(1)))
#define AS_SHARED __attribute__((address_space(3)))

#if __has_builtin(__builtin_amdgcn_global_load_async_to_lds_b128)
#define HAVE_ASYNC_LDS 1
#else
#define HAVE_ASYNC_LDS 0
#endif

__device__ __forceinline__ void wait_async_zero() {
#if __has_builtin(__builtin_amdgcn_s_wait_asynccnt)
    __builtin_amdgcn_s_wait_asynccnt(0);
#else
    asm volatile("s_wait_asynccnt 0x0" ::: "memory");
#endif
}

__device__ __forceinline__ unsigned short f2bf_bits(float f) {
    unsigned u = __float_as_uint(f);
    unsigned r = u + 0x7FFFu + ((u >> 16) & 1u);   // round-to-nearest-even
    return (unsigned short)(r >> 16);
}
__device__ __forceinline__ float siluf(float x) { return x / (1.f + __expf(-x)); }

struct __align__(16) U4 { unsigned int x, y, z, w; };

// ---------------- generic elementwise kernels ----------------
__global__ void embed_kernel(float* __restrict__ h, const int* __restrict__ ids,
                             const float* __restrict__ emb) {
    int idx = blockIdx.x * 256 + threadIdx.x;
    if (idx >= T_TOK * DM) return;
    int t = idx / DM, d = idx - t * DM;
    h[idx] = emb[(size_t)ids[t] * DM + d];
}

__global__ void cvt_bf16_kernel(unsigned short* __restrict__ o,
                                const float* __restrict__ x, int n) {
    int idx = blockIdx.x * 256 + threadIdx.x;
    if (idx < n) o[idx] = f2bf_bits(x[idx]);
}

__global__ void add_inplace_kernel(float* __restrict__ a, const float* __restrict__ b, int n) {
    int idx = blockIdx.x * 256 + threadIdx.x;
    if (idx < n) a[idx] += b[idx];
}

__global__ void silu_mul_bf16_kernel(unsigned short* __restrict__ o,
                                     const float* __restrict__ a,
                                     const float* __restrict__ b, int n) {
    int idx = blockIdx.x * 256 + threadIdx.x;
    if (idx < n) o[idx] = f2bf_bits(siluf(a[idx]) * b[idx]);
}

// ---------------- LayerNorm (768) -> bf16 ----------------
__global__ __launch_bounds__(256) void layernorm_bf16_kernel(
    unsigned short* __restrict__ out, const float* __restrict__ x,
    const float* __restrict__ w, const float* __restrict__ b) {
    int t   = blockIdx.x;
    int tid = threadIdx.x;
    const float* xr = x + (size_t)t * DM;
    float v0 = xr[tid], v1 = xr[tid + 256], v2 = xr[tid + 512];
    float s  = v0 + v1 + v2;
    float sq = v0 * v0 + v1 * v1 + v2 * v2;
    for (int off = 16; off; off >>= 1) {
        s  += __shfl_xor(s,  off, 32);
        sq += __shfl_xor(sq, off, 32);
    }
    __shared__ float rs[8], rq[8];
    int lane = tid & 31, wid = tid >> 5;
    if (lane == 0) { rs[wid] = s; rq[wid] = sq; }
    __syncthreads();
    s = 0.f; sq = 0.f;
    #pragma unroll
    for (int i = 0; i < 8; ++i) { s += rs[i]; sq += rq[i]; }
    float mu  = s * (1.f / DM);
    float var = sq * (1.f / DM) - mu * mu;
    float inv = rsqrtf(var + EPS_F);
    unsigned short* o = out + (size_t)t * DM;
    o[tid]       = f2bf_bits((v0 - mu) * inv * w[tid]       + b[tid]);
    o[tid + 256] = f2bf_bits((v1 - mu) * inv * w[tid + 256] + b[tid + 256]);
    o[tid + 512] = f2bf_bits((v2 - mu) * inv * w[tid + 512] + b[tid + 512]);
}

// ---------------- dt = softplus(raw + bias) ----------------
__global__ void dt_softplus_kernel(float* __restrict__ dt, const float* __restrict__ zx,
                                   const float* __restrict__ dtb) {
    int idx = blockIdx.x * 256 + threadIdx.x;
    if (idx >= T_TOK * NH) return;
    int t = idx / NH, hh = idx - t * NH;
    float v = zx[(size_t)t * DIP + DI + CD + hh] + dtb[hh];
    dt[idx] = (v > 20.f) ? v : log1pf(__expf(v));
}

// ---------------- causal depthwise conv(4) + SiLU ----------------
__global__ void conv_silu_kernel(float* __restrict__ xbc, const float* __restrict__ zx,
                                 const float* __restrict__ cw, const float* __restrict__ cb) {
    int idx = blockIdx.x * 256 + threadIdx.x;
    if (idx >= T_TOK * CD) return;
    int t = idx / CD, c = idx - t * CD;
    int b = t / SEQ, l = t - b * SEQ;
    float acc = cb[c];
    #pragma unroll
    for (int k = 0; k < DCONV; ++k) {
        int ll = l - (DCONV - 1) + k;
        if (ll >= 0)
            acc += cw[c * DCONV + k] * zx[(size_t)(b * SEQ + ll) * DIP + DI + c];
    }
    xbc[idx] = siluf(acc);
}

// ---------------- sequential SSM scan ----------------
// one block per (batch, head); 64x128 state in registers (32 f32 / lane)
__global__ __launch_bounds__(256) void ssm_scan_kernel(
    float* __restrict__ y, const float* __restrict__ xbc,
    const float* __restrict__ dtb, const float* __restrict__ A_log,
    const float* __restrict__ Dp) {
    int b  = blockIdx.x / NH, hh = blockIdx.x - b * NH;
    int tid = threadIdx.x, lane = tid & 31;
    float Ac = -__expf(A_log[hh]);
    float Dc = Dp[hh];
    int n     = tid & (DS_N - 1);   // state column owned by this lane
    int pbase = tid >> 7;           // 0 or 1; rows p = pbase + 2j
    float st[32];
    #pragma unroll
    for (int j = 0; j < 32; ++j) st[j] = 0.f;
    __shared__ float xs[HD], Bs[DS_N], Cs[DS_N], ys[HD];
    for (int l = 0; l < SEQ; ++l) {
        int t = b * SEQ + l;
        const float* row = xbc + (size_t)t * CD;
        if (tid < DS_N) Bs[tid] = row[DI + tid];
        else            Cs[tid - DS_N] = row[DI + DS_N + (tid - DS_N)];
        if (tid < HD) { xs[tid] = row[hh * HD + tid]; ys[tid] = 0.f; }
        __syncthreads();
        float dt = dtb[(size_t)t * NH + hh];
        float dA = __expf(dt * Ac);
        float dB = dt * Bs[n];
        float Cn = Cs[n];
        #pragma unroll
        for (int j = 0; j < 32; ++j) {
            int p = pbase + 2 * j;
            float s = st[j] * dA + dB * xs[p];
            st[j] = s;
            float part = s * Cn;
            for (int off = 16; off; off >>= 1) part += __shfl_xor(part, off, 32);
            if (lane == 0) atomicAdd(&ys[p], part);
        }
        __syncthreads();
        if (tid < HD) y[(size_t)t * DI + hh * HD + tid] = ys[tid] + Dc * xs[tid];
        __syncthreads();
    }
}

// ---------------- gated RMSNorm (y * silu(z)) -> bf16 ----------------
__global__ __launch_bounds__(256) void gated_rmsnorm_bf16_kernel(
    unsigned short* __restrict__ out, const float* __restrict__ y,
    const float* __restrict__ zx, const float* __restrict__ rw) {
    int t = blockIdx.x, tid = threadIdx.x;
    const float* yr = y  + (size_t)t * DI;
    const float* zr = zx + (size_t)t * DIP;
    float vals[6];
    float sq = 0.f;
    #pragma unroll
    for (int j = 0; j < 6; ++j) {
        int e = tid + 256 * j;
        float v = yr[e] * siluf(zr[e]);
        vals[j] = v;
        sq += v * v;
    }
    for (int off = 16; off; off >>= 1) sq += __shfl_xor(sq, off, 32);
    __shared__ float rq[8];
    int lane = tid & 31, wid = tid >> 5;
    if (lane == 0) rq[wid] = sq;
    __syncthreads();
    sq = 0.f;
    #pragma unroll
    for (int i = 0; i < 8; ++i) sq += rq[i];
    float inv = rsqrtf(sq * (1.f / DI) + EPS_F);
    unsigned short* o = out + (size_t)t * DI;
    #pragma unroll
    for (int j = 0; j < 6; ++j) {
        int e = tid + 256 * j;
        o[e] = f2bf_bits(vals[j] * inv * rw[e]);
    }
}

// ---------------- WMMA bf16 GEMM: C[M,N] = A[M,K] * B[N,K]^T ----------------
#define MT 128
#define NT 64
#define KT 32
#define LDS_S 48   // ushort row stride (32 data + 16 pad) -> 96B rows, 16B aligned

__global__ __launch_bounds__(256) void gemm_bf16_wmma_kernel(
    float* __restrict__ C, const unsigned short* __restrict__ A,
    const unsigned short* __restrict__ B, int M, int N, int K) {
    __shared__ __align__(16) unsigned short As[MT * LDS_S];
    __shared__ __align__(16) unsigned short Bs[NT * LDS_S];
    const int tid  = threadIdx.x;
    const int lane = tid & 31, wid = tid >> 5;
    const int bm = blockIdx.y * MT;
    const int bn = blockIdx.x * NT;
    const int wm = (wid >> 1) * 32;   // 4 wave rows -> 128
    const int wn = (wid & 1) * 32;    // 2 wave cols -> 64
    f32x8 acc[2][2] = {};
    const int hf = lane >> 4, l16 = lane & 15;

    // Edge blocks in N: pre-zero B tile once; OOB rows are never overwritten.
    if (bn + NT > N) {
        for (int u = tid; u < NT * LDS_S / 2; u += 256)
            ((unsigned int*)Bs)[u] = 0u;
        __syncthreads();
    }

    const int brow = tid >> 2, bq = tid & 3;  // B tile: 64 rows x 4 quads of 16B
    const int bvalid = (bn + brow < N);

    for (int k0 = 0; k0 < K; k0 += KT) {
#if HAVE_ASYNC_LDS
        // ---- CDNA5 async copy: global -> LDS, no VGPR staging (ASYNCcnt) ----
        #pragma unroll
        for (int it = 0; it < 2; ++it) {          // A tile: 128 rows x 4 quads
            int v = tid + 256 * it;
            int r = v >> 2, q = v & 3;
            const unsigned short* gp = A + (size_t)(bm + r) * K + k0 + q * 8;
            __builtin_amdgcn_global_load_async_to_lds_b128(
                (AS_GLOBAL i32x4*)gp, (AS_SHARED i32x4*)&As[r * LDS_S + q * 8], 0, 0);
        }
        if (bvalid) {
            const unsigned short* gp = B + (size_t)(bn + brow) * K + k0 + bq * 8;
            __builtin_amdgcn_global_load_async_to_lds_b128(
                (AS_GLOBAL i32x4*)gp, (AS_SHARED i32x4*)&Bs[brow * LDS_S + bq * 8], 0, 0);
        }
        wait_async_zero();
#else
        // ---- fallback: vectorized b128 via VGPRs ----
        #pragma unroll
        for (int it = 0; it < 2; ++it) {
            int v = tid + 256 * it;
            int r = v >> 2, q = v & 3;
            U4 val = *(const U4*)(A + (size_t)(bm + r) * K + k0 + q * 8);
            *(U4*)&As[r * LDS_S + q * 8] = val;
        }
        if (bvalid) {
            U4 val = *(const U4*)(B + (size_t)(bn + brow) * K + k0 + bq * 8);
            *(U4*)&Bs[brow * LDS_S + bq * 8] = val;
        }
#endif
        __syncthreads();

        union Frag { bf16x16 v; unsigned int u[8]; };
        Frag af[2], bf[2];
        #pragma unroll
        for (int i = 0; i < 2; ++i) {
            int row = wm + 16 * i + l16;
            #pragma unroll
            for (int v = 0; v < 8; ++v) {
                int kb = (v < 4 ? 2 * v : 16 + 2 * (v - 4)) + 8 * hf; // A: K pairs, 8*half split
                af[i].u[v] = *(const unsigned int*)&As[row * LDS_S + kb];
            }
        }
        #pragma unroll
        for (int j = 0; j < 2; ++j) {
            int col = wn + 16 * j + l16;
            #pragma unroll
            for (int v = 0; v < 8; ++v) {
                int kb = (v < 4 ? 2 * v : 16 + 2 * (v - 4)) + 8 * hf; // B: K pairs, 8*half split
                bf[j].u[v] = *(const unsigned int*)&Bs[col * LDS_S + kb];
            }
        }
        #pragma unroll
        for (int i = 0; i < 2; ++i)
            #pragma unroll
            for (int j = 0; j < 2; ++j)
                acc[i][j] = __builtin_amdgcn_wmma_f32_16x16x32_bf16(
                    false, af[i].v, false, bf[j].v, (short)0, acc[i][j], false, false);
        __syncthreads();
    }

    // store: C/D layout -> VGPR r: row = 16i + r + 8*half, col = 16j + lane%16
    #pragma unroll
    for (int i = 0; i < 2; ++i)
        #pragma unroll
        for (int j = 0; j < 2; ++j) {
            int col = bn + wn + 16 * j + l16;
            if (col < N) {
                #pragma unroll
                for (int r = 0; r < 8; ++r) {
                    int row = bm + wm + 16 * i + r + 8 * hf;
                    C[(size_t)row * N + col] = acc[i][j][r];
                }
            }
        }
}

// ---------------- host orchestration ----------------
extern "C" void kernel_launch(void* const* d_in, const int* in_sizes, int n_in,
                              void* d_out, int out_size, void* d_ws, size_t ws_size,
                              hipStream_t stream) {
    (void)in_sizes; (void)n_in; (void)out_size; (void)ws_size;
    const int*   ids      = (const int*)d_in[0];
    const float* emb      = (const float*)d_in[1];
    const float* ln_w     = (const float*)d_in[2];
    const float* ln_b     = (const float*)d_in[3];
    const float* in_w     = (const float*)d_in[4];
    const float* conv_w   = (const float*)d_in[5];
    const float* conv_b   = (const float*)d_in[6];
    const float* dt_bias  = (const float*)d_in[7];
    const float* A_log    = (const float*)d_in[8];
    const float* Dp       = (const float*)d_in[9];
    const float* rms_w    = (const float*)d_in[10];
    const float* out_w    = (const float*)d_in[11];
    const float* mlp_ln_w = (const float*)d_in[12];
    const float* mlp_ln_b = (const float*)d_in[13];
    const float* w1       = (const float*)d_in[14];
    const float* w2       = (const float*)d_in[15];
    const float* w3       = (const float*)d_in[16];
    const float* normf_w  = (const float*)d_in[17];
    const float* normf_b  = (const float*)d_in[18];
    const float* lm_w     = (const float*)d_in[19];
    float* logits = (float*)d_out;

    char* ws = (char*)d_ws;
    auto au = [](size_t x) { return (x + 255) & ~(size_t)255; };
    size_t off = 0;
    float* h             = (float*)(ws + off);          off = au(off + (size_t)T_TOK * DM * 4);
    unsigned short* hnbf = (unsigned short*)(ws + off); off = au(off + (size_t)T_TOK * DM * 2);
    size_t arena = off;                                 off = au(off + (size_t)2 * T_TOK * FF * 4);
    float* zx  = (float*)(ws + arena);                       // T*DIP f32 (54.9MB) fits in arena
    float* ffa = (float*)(ws + arena);                       // phase B reuse
    float* ffb = (float*)(ws + arena + (size_t)T_TOK * FF * 4);
    float* xbc           = (float*)(ws + off);          off = au(off + (size_t)T_TOK * CD * 4);
    float* dt            = (float*)(ws + off);          off = au(off + (size_t)T_TOK * NH * 4);
    float* ybuf          = (float*)(ws + off);          off = au(off + (size_t)T_TOK * DI * 4);
    unsigned short* gbf  = (unsigned short*)(ws + off); off = au(off + (size_t)T_TOK * FF * 2);
    unsigned short* wbf  = (unsigned short*)(ws + off); off = au(off + (size_t)VOCAB * DM * 2);

    const int TPB = 256;
    auto g1 = [](int n) { return dim3((unsigned)((n + 255) / 256)); };
    dim3 ggIP((DIP + NT - 1) / NT, T_TOK / MT);
    dim3 ggDM((DM  + NT - 1) / NT, T_TOK / MT);
    dim3 ggFF((FF  + NT - 1) / NT, T_TOK / MT);
    dim3 ggLM((VOCAB + NT - 1) / NT, T_TOK / MT);

    embed_kernel<<<g1(T_TOK * DM), TPB, 0, stream>>>(h, ids, emb);

    for (int i = 0; i < NLAYERS; ++i) {
        // ---- mixer ----
        layernorm_bf16_kernel<<<T_TOK, TPB, 0, stream>>>(hnbf, h, ln_w + i * DM, ln_b + i * DM);
        cvt_bf16_kernel<<<g1(DIP * DM), TPB, 0, stream>>>(wbf, in_w + (size_t)i * DIP * DM, DIP * DM);
        gemm_bf16_wmma_kernel<<<ggIP, TPB, 0, stream>>>(zx, hnbf, wbf, T_TOK, DIP, DM);
        dt_softplus_kernel<<<g1(T_TOK * NH), TPB, 0, stream>>>(dt, zx, dt_bias + i * NH);
        conv_silu_kernel<<<g1(T_TOK * CD), TPB, 0, stream>>>(
            xbc, zx, conv_w + (size_t)i * CD * DCONV, conv_b + (size_t)i * CD);
        ssm_scan_kernel<<<BATCH * NH, TPB, 0, stream>>>(ybuf, xbc, dt, A_log + i * NH, Dp + i * NH);
        gated_rmsnorm_bf16_kernel<<<T_TOK, TPB, 0, stream>>>(gbf, ybuf, zx, rms_w + (size_t)i * DI);
        cvt_bf16_kernel<<<g1(DM * DI), TPB, 0, stream>>>(wbf, out_w + (size_t)i * DM * DI, DM * DI);
        gemm_bf16_wmma_kernel<<<ggDM, TPB, 0, stream>>>(ybuf, gbf, wbf, T_TOK, DM, DI);
        add_inplace_kernel<<<g1(T_TOK * DM), TPB, 0, stream>>>(h, ybuf, T_TOK * DM);
        // ---- MLP ----
        layernorm_bf16_kernel<<<T_TOK, TPB, 0, stream>>>(hnbf, h, mlp_ln_w + i * DM, mlp_ln_b + i * DM);
        cvt_bf16_kernel<<<g1(FF * DM), TPB, 0, stream>>>(wbf, w1 + (size_t)i * FF * DM, FF * DM);
        gemm_bf16_wmma_kernel<<<ggFF, TPB, 0, stream>>>(ffa, hnbf, wbf, T_TOK, FF, DM);
        cvt_bf16_kernel<<<g1(FF * DM), TPB, 0, stream>>>(wbf, w2 + (size_t)i * FF * DM, FF * DM);
        gemm_bf16_wmma_kernel<<<ggFF, TPB, 0, stream>>>(ffb, hnbf, wbf, T_TOK, FF, DM);
        silu_mul_bf16_kernel<<<g1(T_TOK * FF), TPB, 0, stream>>>(gbf, ffa, ffb, T_TOK * FF);
        cvt_bf16_kernel<<<g1(DM * FF), TPB, 0, stream>>>(wbf, w3 + (size_t)i * DM * FF, DM * FF);
        gemm_bf16_wmma_kernel<<<ggDM, TPB, 0, stream>>>(ybuf, gbf, wbf, T_TOK, DM, FF);
        add_inplace_kernel<<<g1(T_TOK * DM), TPB, 0, stream>>>(h, ybuf, T_TOK * DM);
    }

    // ---- final LN + lm_head ----
    layernorm_bf16_kernel<<<T_TOK, TPB, 0, stream>>>(hnbf, h, normf_w, normf_b);
    cvt_bf16_kernel<<<g1(VOCAB * DM), TPB, 0, stream>>>(wbf, lm_w, (int)(VOCAB * DM));
    gemm_bf16_wmma_kernel<<<ggLM, TPB, 0, stream>>>(logits, hnbf, wbf, T_TOK, VOCAB, DM);
}
